// model_1735166787673
// MI455X (gfx1250) — matrix operand order repaired and checked by reading, not compiled, not run on hardware
//
#include <hip/hip_runtime.h>
#include <math.h>

// ---- model constants ----
#define BB   16
#define SS   512
#define LL   12
#define HH   12
#define DD   768
#define FF   3072
#define DHH  64
#define NLL  9
#define MTOK (BB * SS)   // 8192 tokens

typedef __attribute__((ext_vector_type(16))) __bf16 v16bf;
typedef __attribute__((ext_vector_type(8)))  __bf16 v8bf;
typedef __attribute__((ext_vector_type(8)))  float  v8f;

__device__ __forceinline__ v8f wmma_bf16(v16bf a, v16bf b, v8f c) {
  // 8 args: (neg_a, A, neg_b, B, c_mod, C, reuse_a, reuse_b)
  return __builtin_amdgcn_wmma_f32_16x16x32_bf16(false, a, false, b, (short)0, c, false, false);
}

__device__ __forceinline__ float rowmax16(float v) {
  v = fmaxf(v, __shfl_xor(v, 1, 32));
  v = fmaxf(v, __shfl_xor(v, 2, 32));
  v = fmaxf(v, __shfl_xor(v, 4, 32));
  v = fmaxf(v, __shfl_xor(v, 8, 32));
  return v;
}
__device__ __forceinline__ float rowsum16(float v) {
  v += __shfl_xor(v, 1, 32);
  v += __shfl_xor(v, 2, 32);
  v += __shfl_xor(v, 4, 32);
  v += __shfl_xor(v, 8, 32);
  return v;
}

// LDS byte offset of a __shared__ object = low 32 bits of its generic address
// (shared aperture occupies addr[63:32] only; see CDNA5 aperture check).
__device__ __forceinline__ unsigned lds_off(const void* p) {
  return (unsigned)(unsigned long long)(size_t)p;
}

// ---------------- fp32 -> bf16 convert ----------------
__global__ __launch_bounds__(256) void cvt_kernel(const float* __restrict__ src,
                                                  __bf16* __restrict__ dst, int n) {
  int i = blockIdx.x * 256 + threadIdx.x;
  if (i < n) dst[i] = (__bf16)src[i];
}

// ---------------- embeddings: y = word + pos + type0 ----------------
__global__ __launch_bounds__(256) void embed_kernel(const long long* __restrict__ ids,
                                                    const float* __restrict__ we,
                                                    const float* __restrict__ pe,
                                                    const float* __restrict__ te,
                                                    float* __restrict__ y) {
  int tok = blockIdx.x;
  int s = tok & (SS - 1);
  long long id = ids[tok];
  const float* wrow = we + (size_t)id * DD;
  const float* prow = pe + (size_t)s * DD;
  for (int i = threadIdx.x; i < DD; i += 256)
    y[(size_t)tok * DD + i] = wrow[i] + prow[i] + te[i];
}

// ---------------- layernorm over 768, writes fp32 + bf16 ----------------
__global__ __launch_bounds__(256) void ln_kernel(const float* __restrict__ y,
                                                 const float* __restrict__ g,
                                                 const float* __restrict__ b,
                                                 float* __restrict__ xf,
                                                 __bf16* __restrict__ xb) {
  __shared__ float red[256];
  int tok = blockIdx.x;
  const float* row = y + (size_t)tok * DD;
  int t = threadIdx.x;
  float v0 = row[t], v1 = row[t + 256], v2 = row[t + 512];

  red[t] = v0 + v1 + v2;
  __syncthreads();
  for (int off = 128; off > 0; off >>= 1) {
    if (t < off) red[t] += red[t + off];
    __syncthreads();
  }
  float mean = red[0] * (1.0f / DD);
  __syncthreads();

  float d0 = v0 - mean, d1 = v1 - mean, d2 = v2 - mean;
  red[t] = d0 * d0 + d1 * d1 + d2 * d2;
  __syncthreads();
  for (int off = 128; off > 0; off >>= 1) {
    if (t < off) red[t] += red[t + off];
    __syncthreads();
  }
  float inv = rsqrtf(red[0] * (1.0f / DD) + 1e-12f);

  size_t base = (size_t)tok * DD;
  float o0 = d0 * inv * g[t] + b[t];
  float o1 = d1 * inv * g[t + 256] + b[t + 256];
  float o2 = d2 * inv * g[t + 512] + b[t + 512];
  xf[base + t] = o0;        xb[base + t] = (__bf16)o0;
  xf[base + t + 256] = o1;  xb[base + t + 256] = (__bf16)o1;
  xf[base + t + 512] = o2;  xb[base + t + 512] = (__bf16)o2;
}

// ---------------- generic WMMA GEMM: Y = A[MxK] @ W[KxN] + bias ----------------
// block tile 128x64, 8 waves as 4x2, each wave 32x32 (2x2 wmma tiles)
// A tile staged global->LDS via the CDNA5 async engine (ASYNCcnt),
// W tile staged transposed so B-fragments are contiguous 32B LDS loads.
// modes: 0 = bf16 out, Q/K head layout [B,H,S,DH]
//        1 = bf16 out, V transposed   [B,H,DH,S]
//        2 = fp32 out + residual add  (outf[m*N+n] = acc + bias + resid)
//        3 = bf16 out, exact GELU     (FFN1)
__global__ __launch_bounds__(256) void gemm_wmma(const __bf16* __restrict__ A,
                                                 const __bf16* __restrict__ W,
                                                 const float* __restrict__ bias,
                                                 int M, int K, int N, int mode,
                                                 float* __restrict__ outf,
                                                 const float* __restrict__ resid,
                                                 __bf16* __restrict__ outb) {
  __shared__ __bf16 sA[128 * 32];   // [m_local][k]  8KB
  __shared__ __bf16 sW[64 * 32];    // [n_local][k] (transposed) 4KB

  const int nBlocks = N >> 6;
  const int bm = (blockIdx.x / nBlocks) << 7;
  const int bn = (blockIdx.x % nBlocks) << 6;
  const int t = threadIdx.x;
  const int lane = t & 31;
  const int w = t >> 5;
  const int wm = w >> 1, wn = w & 1;
  const int half = lane >> 4;
  const int l15 = lane & 15;

  const int arow = t >> 1, ach = (t & 1) << 4;          // A staging: 32B/thread
  const int wkk = t >> 3, wnn = (t & 7) << 3;           // W staging: 8 elems

  const unsigned aLds = lds_off(&sA[arow * 32 + ach]);

  v8f acc[2][2] = {};

  for (int k0 = 0; k0 < K; k0 += 32) {
    // stage A tile: async global->LDS copy, 2 x b128 per thread.
    // INST_OFFSET applies to both the LDS and global addresses (ISA 08 §4.4).
    {
      unsigned long long gA =
          (unsigned long long)(size_t)&A[(size_t)(bm + arow) * K + k0 + ach];
      asm volatile("global_load_async_to_lds_b128 %0, %1, off"
                   :: "v"(aLds), "v"(gA) : "memory");
      asm volatile("global_load_async_to_lds_b128 %0, %1, off offset:16"
                   :: "v"(aLds), "v"(gA) : "memory");
      if (k0 + 32 < K) __builtin_prefetch(&A[(size_t)(bm + arow) * K + k0 + 32 + ach], 0, 3);
    }
    // stage W tile transposed: sW[n][k]
    {
      v8bf x = reinterpret_cast<const v8bf*>(&W[(size_t)(k0 + wkk) * N + bn + wnn])[0];
#pragma unroll
      for (int j = 0; j < 8; ++j) sW[(wnn + j) * 32 + wkk] = x[j];
    }
    asm volatile("s_wait_asynccnt 0x0" ::: "memory");
    __syncthreads();

    // A frags: lane m = l15; k(e) = half*16 + e
    v16bf af[2];
#pragma unroll
    for (int i = 0; i < 2; ++i) {
      int row = wm * 32 + i * 16 + l15;
      v8bf lo = reinterpret_cast<const v8bf*>(&sA[row * 32 + half * 16])[0];
      v8bf hi = reinterpret_cast<const v8bf*>(&sA[row * 32 + half * 16 + 8])[0];
      v16bf a;
#pragma unroll
      for (int e = 0; e < 8; ++e) { a[e] = lo[e]; a[e + 8] = hi[e]; }
      af[i] = a;
    }
    // B frags: lane n = l15; k(e) = half*16 + e (contiguous in transposed sW)
    v16bf bfg[2];
#pragma unroll
    for (int j = 0; j < 2; ++j) {
      int col = wn * 32 + j * 16 + l15;
      bfg[j] = reinterpret_cast<const v16bf*>(&sW[col * 32 + half * 16])[0];
    }
#pragma unroll
    for (int i = 0; i < 2; ++i)
#pragma unroll
      for (int j = 0; j < 2; ++j)
        acc[i][j] = wmma_bf16(af[i], bfg[j], acc[i][j]);
    __syncthreads();
  }

  // epilogue: C/D layout m = r + 8*half, n = l15
#pragma unroll
  for (int i = 0; i < 2; ++i) {
#pragma unroll
    for (int j = 0; j < 2; ++j) {
      int ncol = bn + wn * 32 + j * 16 + l15;
      float bv = bias[ncol];
#pragma unroll
      for (int r = 0; r < 8; ++r) {
        int mrow = bm + wm * 32 + i * 16 + r + half * 8;
        float v = acc[i][j][r] + bv;
        if (mode == 0) {
          int b = mrow >> 9, s = mrow & (SS - 1), h = ncol >> 6, d = ncol & 63;
          outb[((size_t)(b * HH + h) * SS + s) * DHH + d] = (__bf16)v;
        } else if (mode == 1) {
          int b = mrow >> 9, s = mrow & (SS - 1), h = ncol >> 6, d = ncol & 63;
          outb[((size_t)(b * HH + h) * DHH + d) * SS + s] = (__bf16)v;
        } else if (mode == 2) {
          size_t idx = (size_t)mrow * N + ncol;
          outf[idx] = v + resid[idx];
        } else {
          float gl = 0.5f * v * (1.0f + erff(v * 0.70710678118654752f));
          outb[(size_t)mrow * N + ncol] = (__bf16)gl;
        }
      }
    }
  }
}

// ---------------- flash attention: one wave per 16-row Q tile ----------------
// grid = B * H * (S/128); 8 waves/block cover 128 q rows
__global__ __launch_bounds__(256) void attn_kernel(const __bf16* __restrict__ qg,
                                                   const __bf16* __restrict__ kg,
                                                   const __bf16* __restrict__ vtg,
                                                   const int* __restrict__ amask,
                                                   __bf16* __restrict__ ctx) {
  __shared__ __bf16 sP[8][16 * 32];   // per-wave P bounce buffer (8KB)

  const int t = threadIdx.x;
  const int lane = t & 31, w = t >> 5;
  const int half = lane >> 4, l15 = lane & 15;
  const int qb = blockIdx.x & 3;
  const int h = (blockIdx.x >> 2) % HH;
  const int b = blockIdx.x / (4 * HH);
  const int qbase = qb * 128 + w * 16;

  const __bf16* qp = qg + (size_t)(b * HH + h) * SS * DHH;
  const __bf16* kp = kg + (size_t)(b * HH + h) * SS * DHH;
  const __bf16* vp = vtg + (size_t)(b * HH + h) * DHH * SS;

  // Q A-frags (dh 0..31, 32..63), loaded once
  v16bf qa[2];
#pragma unroll
  for (int f = 0; f < 2; ++f) {
    const __bf16* qrow = &qp[(size_t)(qbase + l15) * DHH + f * 32];
    v8bf lo = reinterpret_cast<const v8bf*>(qrow + half * 16)[0];
    v8bf hi = reinterpret_cast<const v8bf*>(qrow + half * 16 + 8)[0];
    v16bf a;
#pragma unroll
    for (int e = 0; e < 8; ++e) { a[e] = lo[e]; a[e + 8] = hi[e]; }
    qa[f] = a;
  }

  float mrun[8], lrun[8];
  v8f o[4] = {};
#pragma unroll
  for (int r = 0; r < 8; ++r) { mrun[r] = -3.0e38f; lrun[r] = 0.0f; }

  for (int kk = 0; kk < SS; kk += 32) {
    // S = Q @ K^T for two 16-key tiles
    v8f s0 = {}, s1 = {};
#pragma unroll
    for (int f = 0; f < 2; ++f) {
      v16bf kb0 = reinterpret_cast<const v16bf*>(
          &kp[(size_t)(kk + l15) * DHH + f * 32 + half * 16])[0];
      v16bf kb1 = reinterpret_cast<const v16bf*>(
          &kp[(size_t)(kk + 16 + l15) * DHH + f * 32 + half * 16])[0];
      s0 = wmma_bf16(qa[f], kb0, s0);
      s1 = wmma_bf16(qa[f], kb1, s1);
    }

    float bias0 = (amask[b * SS + kk + l15] > 0) ? 0.0f : -1e4f;
    float bias1 = (amask[b * SS + kk + 16 + l15] > 0) ? 0.0f : -1e4f;

    float p0[8], p1[8];
#pragma unroll
    for (int r = 0; r < 8; ++r) {
      float a = s0[r] * 0.125f + bias0;   // 1/sqrt(64)
      float c = s1[r] * 0.125f + bias1;
      float tmax = rowmax16(fmaxf(a, c));
      float mnew = fmaxf(mrun[r], tmax);
      float sf = expf(mrun[r] - mnew);
      float e0 = expf(a - mnew), e1 = expf(c - mnew);
      float rs = rowsum16(e0 + e1);
      lrun[r] = lrun[r] * sf + rs;
      mrun[r] = mnew;
#pragma unroll
      for (int td = 0; td < 4; ++td) o[td][r] *= sf;
      p0[r] = e0; p1[r] = e1;
    }

    // C-layout -> A-layout transpose of P via LDS
#pragma unroll
    for (int r = 0; r < 8; ++r) {
      sP[w][(r + 8 * half) * 32 + l15] = (__bf16)p0[r];
      sP[w][(r + 8 * half) * 32 + 16 + l15] = (__bf16)p1[r];
    }
    __syncthreads();

    v16bf pa;
    {
      v8bf lo = reinterpret_cast<const v8bf*>(&sP[w][l15 * 32 + half * 16])[0];
      v8bf hi = reinterpret_cast<const v8bf*>(&sP[w][l15 * 32 + half * 16 + 8])[0];
#pragma unroll
      for (int e = 0; e < 8; ++e) { pa[e] = lo[e]; pa[e + 8] = hi[e]; }
    }
    // O += P @ V (V pre-transposed [DH,S] so B frags are contiguous)
#pragma unroll
    for (int td = 0; td < 4; ++td) {
      v16bf vb = reinterpret_cast<const v16bf*>(
          &vp[(size_t)(td * 16 + l15) * SS + kk + half * 16])[0];
      o[td] = wmma_bf16(pa, vb, o[td]);
    }
    __syncthreads();
  }

  // finalize: ctx token-major bf16 [B*S, D]
#pragma unroll
  for (int td = 0; td < 4; ++td) {
#pragma unroll
    for (int r = 0; r < 8; ++r) {
      int q = qbase + r + 8 * half;
      float v = o[td][r] / lrun[r];
      ctx[(size_t)(b * SS + q) * DD + h * DHH + td * 16 + l15] = (__bf16)v;
    }
  }
}

// ---------------- classifier + softmax over 9 labels ----------------
__global__ __launch_bounds__(256) void cls_kernel(const float* __restrict__ x,
                                                  const float* __restrict__ Wc,
                                                  const float* __restrict__ bc,
                                                  float* __restrict__ out) {
  __shared__ float sW[DD * NLL];
  for (int i = threadIdx.x; i < DD * NLL; i += 256) sW[i] = Wc[i];
  __syncthreads();

  int tok = blockIdx.x * 256 + threadIdx.x;
  const float* row = x + (size_t)tok * DD;
  float acc[NLL];
#pragma unroll
  for (int c = 0; c < NLL; ++c) acc[c] = bc[c];
  for (int d = 0; d < DD; ++d) {
    float xv = row[d];
#pragma unroll
    for (int c = 0; c < NLL; ++c) acc[c] += xv * sW[d * NLL + c];
  }
  float mx = acc[0];
#pragma unroll
  for (int c = 1; c < NLL; ++c) mx = fmaxf(mx, acc[c]);
  float s = 0.0f;
#pragma unroll
  for (int c = 0; c < NLL; ++c) { acc[c] = expf(acc[c] - mx); s += acc[c]; }
  float inv = 1.0f / s;
#pragma unroll
  for (int c = 0; c < NLL; ++c) out[(size_t)tok * NLL + c] = acc[c] * inv;
}

// ---------------- host orchestration ----------------
extern "C" void kernel_launch(void* const* d_in, const int* in_sizes, int n_in,
                              void* d_out, int out_size, void* d_ws, size_t ws_size,
                              hipStream_t stream) {
  const long long* ids = (const long long*)d_in[0];
  const int* amask = (const int*)d_in[1];
  const float* we = (const float*)d_in[2];
  const float* pe = (const float*)d_in[3];
  const float* te = (const float*)d_in[4];
  const float* eg = (const float*)d_in[5];
  const float* eb = (const float*)d_in[6];
  const float* Wq = (const float*)d_in[7];
  const float* bq = (const float*)d_in[8];
  const float* Wk = (const float*)d_in[9];
  const float* bk = (const float*)d_in[10];
  const float* Wv = (const float*)d_in[11];
  const float* bv = (const float*)d_in[12];
  const float* Wo = (const float*)d_in[13];
  const float* bo = (const float*)d_in[14];
  const float* ln1g = (const float*)d_in[15];
  const float* ln1b = (const float*)d_in[16];
  const float* W1 = (const float*)d_in[17];
  const float* b1 = (const float*)d_in[18];
  const float* W2 = (const float*)d_in[19];
  const float* b2 = (const float*)d_in[20];
  const float* ln2g = (const float*)d_in[21];
  const float* ln2b = (const float*)d_in[22];
  const float* Wc = (const float*)d_in[23];
  const float* bc = (const float*)d_in[24];

  // workspace carve (256B-aligned chunks)
  char* p = (char*)d_ws;
  auto carve = [&](size_t bytes) -> void* {
    void* r = (void*)p;
    p += (bytes + 255) & ~(size_t)255;
    return r;
  };
  float* xf = (float*)carve((size_t)MTOK * DD * 4);
  float* yf = (float*)carve((size_t)MTOK * DD * 4);
  __bf16* xb = (__bf16*)carve((size_t)MTOK * DD * 2);
  __bf16* qbuf = (__bf16*)carve((size_t)MTOK * DD * 2);
  __bf16* kbuf = (__bf16*)carve((size_t)MTOK * DD * 2);
  __bf16* vbuf = (__bf16*)carve((size_t)MTOK * DD * 2);
  __bf16* ctxb = (__bf16*)carve((size_t)MTOK * DD * 2);
  __bf16* hb = (__bf16*)carve((size_t)MTOK * FF * 2);
  __bf16* wqb = (__bf16*)carve((size_t)DD * DD * 2);
  __bf16* wkb = (__bf16*)carve((size_t)DD * DD * 2);
  __bf16* wvb = (__bf16*)carve((size_t)DD * DD * 2);
  __bf16* wob = (__bf16*)carve((size_t)DD * DD * 2);
  __bf16* w1b = (__bf16*)carve((size_t)DD * FF * 2);
  __bf16* w2b = (__bf16*)carve((size_t)FF * DD * 2);

  const int nDD = DD * DD;        // 589824
  const int nDF = DD * FF;        // 2359296
  const int gDD = (nDD + 255) / 256;
  const int gDF = (nDF + 255) / 256;
  const int gemmDD = (MTOK / 128) * (DD / 64);   // 768 blocks
  const int gemmDF = (MTOK / 128) * (FF / 64);   // 3072 blocks
  const int attnG = BB * HH * (SS / 128);        // 768 blocks

  // embeddings + embedding LN
  embed_kernel<<<MTOK, 256, 0, stream>>>(ids, we, pe, te, yf);
  ln_kernel<<<MTOK, 256, 0, stream>>>(yf, eg, eb, xf, xb);

  for (int l = 0; l < LL; ++l) {
    cvt_kernel<<<gDD, 256, 0, stream>>>(Wq + (size_t)l * nDD, wqb, nDD);
    cvt_kernel<<<gDD, 256, 0, stream>>>(Wk + (size_t)l * nDD, wkb, nDD);
    cvt_kernel<<<gDD, 256, 0, stream>>>(Wv + (size_t)l * nDD, wvb, nDD);
    cvt_kernel<<<gDD, 256, 0, stream>>>(Wo + (size_t)l * nDD, wob, nDD);
    cvt_kernel<<<gDF, 256, 0, stream>>>(W1 + (size_t)l * nDF, w1b, nDF);
    cvt_kernel<<<gDF, 256, 0, stream>>>(W2 + (size_t)l * nDF, w2b, nDF);

    gemm_wmma<<<gemmDD, 256, 0, stream>>>(xb, wqb, bq + l * DD, MTOK, DD, DD, 0,
                                          nullptr, nullptr, qbuf);
    gemm_wmma<<<gemmDD, 256, 0, stream>>>(xb, wkb, bk + l * DD, MTOK, DD, DD, 0,
                                          nullptr, nullptr, kbuf);
    gemm_wmma<<<gemmDD, 256, 0, stream>>>(xb, wvb, bv + l * DD, MTOK, DD, DD, 1,
                                          nullptr, nullptr, vbuf);
    attn_kernel<<<attnG, 256, 0, stream>>>(qbuf, kbuf, vbuf, amask, ctxb);
    gemm_wmma<<<gemmDD, 256, 0, stream>>>(ctxb, wob, bo + l * DD, MTOK, DD, DD, 2,
                                          yf, xf, nullptr);
    ln_kernel<<<MTOK, 256, 0, stream>>>(yf, ln1g + l * DD, ln1b + l * DD, xf, xb);
    gemm_wmma<<<gemmDF, 256, 0, stream>>>(xb, w1b, b1 + l * FF, MTOK, DD, FF, 3,
                                          nullptr, nullptr, hb);
    gemm_wmma<<<gemmDD, 256, 0, stream>>>(hb, w2b, b2 + l * DD, MTOK, FF, DD, 2,
                                          yf, xf, nullptr);
    ln_kernel<<<MTOK, 256, 0, stream>>>(yf, ln2g + l * DD, ln2b + l * DD, xf, xb);
  }

  cls_kernel<<<MTOK / 256, 256, 0, stream>>>(xf, Wc, bc, (float*)d_out);
}
